// CausalSelfAttention_51702816309444
// MI455X (gfx1250) — matrix operand order repaired and checked by reading, not compiled, not run on hardware
//
#include <hip/hip_runtime.h>

// ---------------------------------------------------------------------------
// CDNA5 (gfx1250) causal self-attention, bf16 WMMA everywhere.
// ---------------------------------------------------------------------------

typedef __bf16 bf16_t;
typedef __attribute__((ext_vector_type(16))) __bf16 bf16x16;
typedef __attribute__((ext_vector_type(8)))  __bf16 bf16x8;
typedef __attribute__((ext_vector_type(8)))  float  f32x8;

#define B_  2
#define T_  2048
#define C_  2048
#define H_  16
#define HD_ 128
#define C3_ 6144

__device__ __forceinline__ f32x8 wmma_bf16(bf16x16 a, bf16x16 b, f32x8 c) {
  // (neg_a, A, neg_b, B, c_mod, C, reuse_a, reuse_b)
  return __builtin_amdgcn_wmma_f32_16x16x32_bf16(false, a, false, b, (short)0, c,
                                                 false, false);
}

// A-matrix fragment (16x32 bf16, MxK). For lane-half g (= lane>>4):
//   elements 0..7  -> K = g*16 + e       (contiguous 8 bf16)
//   elements 8..15 -> K = g*16 + e       (contiguous 8 bf16)
// `p` points at (row, K-chunk-base).
__device__ __forceinline__ bf16x16 load_afrag(const bf16_t* p, int g) {
  bf16x8 lo = *(const bf16x8*)(p + g * 16);
  bf16x8 hi = *(const bf16x8*)(p + g * 16 + 8);
  return __builtin_shufflevector(lo, hi, 0, 1, 2, 3, 4, 5, 6, 7,
                                 8, 9, 10, 11, 12, 13, 14, 15);
}

// B-matrix fragment (32x16 bf16, KxN) when the source is stored [N][K]
// row-major: lane holds N = lane&15, elements e -> K = (lane>>4)*16 + e,
// i.e. 16 contiguous bf16. `p` points at (&src[N][ (lane>>4)*16 ]).
__device__ __forceinline__ bf16x16 load_bfrag(const bf16_t* p) {
  bf16x8 lo = *(const bf16x8*)(p);
  bf16x8 hi = *(const bf16x8*)(p + 8);
  return __builtin_shufflevector(lo, hi, 0, 1, 2, 3, 4, 5, 6, 7,
                                 8, 9, 10, 11, 12, 13, 14, 15);
}

// ---------------------------------------------------------------------------
// Tiled GEMM: C[M,N] = A[M,K] @ B[K,N].  A fp32 or bf16 row-major, B fp32
// row-major (weights), C fp32.  BM=BN=128, BK=64, 8 waves/block, each wave
// owns a 64x32 sub-tile = 4x2 fragments, 16 WMMAs per stage per wave.
// Next K-tile is prefetched (global_prefetch_b8) while this one computes.
// ---------------------------------------------------------------------------
template <typename AT>
__global__ __launch_bounds__(256) void gemm_wmma_bf16(
    const AT* __restrict__ A, const float* __restrict__ Bm,
    float* __restrict__ Cm, int M, int N, int K) {
  constexpr int BM = 128, BN = 128, BK = 64;
  constexpr int LDA = BK + 8;  // 72 bf16 = 144B rows (16B-aligned, skewed)
  __shared__ bf16_t As[BM][LDA];    // [m][k]
  __shared__ bf16_t BsT[BN][LDA];   // [n][k]  (transposed stage)

  const int tid  = threadIdx.x;
  const int lane = tid & 31;
  const int w    = tid >> 5;
  const int wm   = (w >> 2) * 64;   // wave row offset in tile
  const int wn   = (w & 3) * 32;    // wave col offset in tile
  const int m0   = blockIdx.y * BM;
  const int n0   = blockIdx.x * BN;
  const int lm   = lane & 15;
  const int lg   = lane >> 4;

  // per-thread staging slots
  const int am = tid >> 1;          // A: row 0..127
  const int ak = (tid & 1) * 32;    //    k-span of 32
  const int bk = tid >> 2;          // B: row 0..63
  const int bn = (tid & 3) * 32;    //    n-span of 32

  f32x8 acc[4][2] = {};

  for (int k0 = 0; k0 < K; k0 += BK) {
    const AT* srcA = A + (size_t)(m0 + am) * K + k0 + ak;
    const float* srcB = Bm + (size_t)(k0 + bk) * N + n0 + bn;

    // stage A tile 128x64: convert to bf16 in regs, 4x b128 LDS stores
#pragma unroll
    for (int j = 0; j < 32; j += 8) {
      bf16x8 t;
#pragma unroll
      for (int e = 0; e < 8; ++e) t[e] = (bf16_t)srcA[j + e];
      *(bf16x8*)&As[am][ak + j] = t;
    }
    // stage B tile 64x128 transposed into BsT[n][k]
#pragma unroll
    for (int j = 0; j < 32; ++j) BsT[bn + j][bk] = (bf16_t)srcB[j];

    // prefetch next K-tile into cache while we compute this one
    if (k0 + BK < K) {
      __builtin_prefetch(srcA + BK, 0, 1);
      __builtin_prefetch(srcB + (size_t)BK * N, 0, 1);
    }
    __syncthreads();

#pragma unroll
    for (int kc = 0; kc < BK; kc += 32) {
      bf16x16 afrag[4];
#pragma unroll
      for (int i = 0; i < 4; ++i)
        afrag[i] = load_afrag(&As[wm + i * 16 + lm][kc], lg);
      bf16x16 bfrag[2];
#pragma unroll
      for (int j = 0; j < 2; ++j)
        bfrag[j] = load_bfrag(&BsT[wn + j * 16 + lm][kc + lg * 16]);
#pragma unroll
      for (int i = 0; i < 4; ++i)
#pragma unroll
        for (int j = 0; j < 2; ++j)
          acc[i][j] = wmma_bf16(afrag[i], bfrag[j], acc[i][j]);
    }
    __syncthreads();
  }

  // C/D fragment layout: lane holds N = lane&15, VGPR v -> M = v + 8*(lane>>4)
#pragma unroll
  for (int i = 0; i < 4; ++i)
#pragma unroll
    for (int j = 0; j < 2; ++j) {
      const int row = m0 + wm + i * 16 + 8 * lg;
      const int col = n0 + wn + j * 16 + lm;
#pragma unroll
      for (int v = 0; v < 8; ++v)
        Cm[(size_t)(row + v) * N + col] = acc[i][j][v];
    }
}

// ---------------------------------------------------------------------------
// RoPE + head split.  qkv fp32 [B,T,3C] -> q,k bf16 [B,H,T,HD] (q pre-scaled
// by 1/sqrt(HD)), v transposed bf16 [B,H,HD,T].
// ---------------------------------------------------------------------------
__global__ __launch_bounds__(256) void rope_split_kernel(
    const float* __restrict__ qkv, const float* __restrict__ cosb,
    const float* __restrict__ sinb, bf16_t* __restrict__ qd,
    bf16_t* __restrict__ kd, bf16_t* __restrict__ vT) {
  const int idx = blockIdx.x * blockDim.x + threadIdx.x;  // (b,t,h)
  if (idx >= B_ * T_ * H_) return;
  const int h = idx % H_;
  const int t = (idx / H_) % T_;
  const int b = idx / (H_ * T_);

  const float* qrow = qkv + (size_t)(b * T_ + t) * C3_ + h * HD_;
  const float* krow = qrow + C_;
  const float* vrow = qrow + 2 * C_;
  const float scale = 0.08838834764831845f;  // 1/sqrt(128)

  const size_t qkbase = ((size_t)(b * H_ + h) * T_ + t) * HD_;
  const size_t vbase  = ((size_t)(b * H_ + h) * HD_) * T_ + t;

#pragma unroll 4
  for (int d = 0; d < HD_ / 2; ++d) {
    const float c = cosb[t * (HD_ / 2) + d];
    const float s = sinb[t * (HD_ / 2) + d];
    const float q1 = qrow[d], q2 = qrow[d + HD_ / 2];
    const float k1 = krow[d], k2 = krow[d + HD_ / 2];
    qd[qkbase + d]           = (bf16_t)((q1 * c - q2 * s) * scale);
    qd[qkbase + d + HD_ / 2] = (bf16_t)((q1 * s + q2 * c) * scale);
    kd[qkbase + d]           = (bf16_t)(k1 * c - k2 * s);
    kd[qkbase + d + HD_ / 2] = (bf16_t)(k1 * s + k2 * c);
  }
#pragma unroll 4
  for (int d = 0; d < HD_; ++d) vT[vbase + (size_t)d * T_] = (bf16_t)vrow[d];
}

// ---------------------------------------------------------------------------
// Flash-style causal attention.  One wave (single-wave workgroup: barriers
// degrade to S_NOP) per (b,h,16-query tile).  Q fragments stay in VGPRs;
// 32 keys per step; online softmax with half-wave shfl_xor row reductions;
// P goes through a 1KB LDS bounce to convert C-layout -> A-layout.
// Next step's K/V rows are prefetched while this step computes.
// ---------------------------------------------------------------------------
__global__ __launch_bounds__(32) void attn_kernel(
    const bf16_t* __restrict__ qd, const bf16_t* __restrict__ kd,
    const bf16_t* __restrict__ vT, bf16_t* __restrict__ yb) {
  __shared__ bf16_t P[16][40];  // 16 q-rows x 32 keys, padded rows (80B)

  const int blk = blockIdx.x;          // 0 .. B*H*(T/16)-1
  const int qt  = blk % (T_ / 16);
  const int bh  = blk / (T_ / 16);
  const int q0  = qt * 16;
  const int lane = threadIdx.x & 31;
  const int lm = lane & 15;
  const int lg = lane >> 4;

  const bf16_t* qbase = qd + (size_t)bh * T_ * HD_;
  const bf16_t* kbase = kd + (size_t)bh * T_ * HD_;
  const bf16_t* vbase = vT + (size_t)bh * HD_ * T_;

  // Q A-fragments for the 4 K-chunks of hd=128 (resident all loop long)
  bf16x16 aq[4];
#pragma unroll
  for (int c = 0; c < 4; ++c)
    aq[c] = load_afrag(qbase + (size_t)(q0 + lm) * HD_ + c * 32, lg);

  f32x8 O[8];
#pragma unroll
  for (int nc = 0; nc < 8; ++nc) O[nc] = f32x8{};
  float mrow[8], lrow[8];
#pragma unroll
  for (int v = 0; v < 8; ++v) { mrow[v] = -3.0e38f; lrow[v] = 0.0f; }

  const int nsteps = (q0 + 16 + 31) / 32;
  for (int ks = 0; ks < nsteps; ++ks) {
    const int kb = ks * 32;

    // prefetch next step's K rows (1 row/lane) and a spread of V rows
    if (ks + 1 < nsteps) {
      __builtin_prefetch(kbase + (size_t)(kb + 32 + lane) * HD_, 0, 1);
      __builtin_prefetch(vbase + (size_t)(lane * 4) * T_ + kb + 32, 0, 1);
    }

    // scores for two 16-key halves: S = Q @ K^T  (K rows are contiguous-d)
    f32x8 s0 = f32x8{}, s1 = f32x8{};
#pragma unroll
    for (int c = 0; c < 4; ++c) {
      bf16x16 bk0 =
          load_bfrag(kbase + (size_t)(kb + lm) * HD_ + c * 32 + lg * 16);
      s0 = wmma_bf16(aq[c], bk0, s0);
    }
#pragma unroll
    for (int c = 0; c < 4; ++c) {
      bf16x16 bk1 =
          load_bfrag(kbase + (size_t)(kb + 16 + lm) * HD_ + c * 32 + lg * 16);
      s1 = wmma_bf16(aq[c], bk1, s1);
    }

    // online softmax, per row v (row M = v + 8*lg lives in one half-wave)
#pragma unroll
    for (int v = 0; v < 8; ++v) {
      const int qr = q0 + v + 8 * lg;
      float s0v = s0[v];
      float s1v = s1[v];
      if (kb + lm > qr) s0v = -3.0e38f;        // causal mask
      if (kb + 16 + lm > qr) s1v = -3.0e38f;
      float mx = fmaxf(s0v, s1v);
#pragma unroll
      for (int off = 8; off >= 1; off >>= 1)
        mx = fmaxf(mx, __shfl_xor(mx, off, 32));
      const float mnew  = fmaxf(mrow[v], mx);
      const float alpha = __expf(mrow[v] - mnew);
      const float p0 = __expf(s0v - mnew);
      const float p1 = __expf(s1v - mnew);
      float rs = p0 + p1;
#pragma unroll
      for (int off = 8; off >= 1; off >>= 1) rs += __shfl_xor(rs, off, 32);
      lrow[v] = lrow[v] * alpha + rs;
      mrow[v] = mnew;
#pragma unroll
      for (int nc = 0; nc < 8; ++nc) O[nc][v] *= alpha;
      P[v + 8 * lg][lm]      = (bf16_t)p0;     // C-layout -> LDS
      P[v + 8 * lg][16 + lm] = (bf16_t)p1;
    }
    __syncthreads();  // single-wave workgroup: S_NOP + dscnt fence

    // P as A-fragment (16x32), V^T gives contiguous B-fragments
    bf16x16 pa = load_afrag(&P[lm][0], lg);
#pragma unroll
    for (int nc = 0; nc < 8; ++nc) {
      bf16x16 bv =
          load_bfrag(vbase + (size_t)(nc * 16 + lm) * T_ + kb + lg * 16);
      O[nc] = wmma_bf16(pa, bv, O[nc]);
    }
    __syncthreads();
  }

  // normalize and scatter into y[B,T,C] (bf16) for the projection GEMM
  const int b = bh >> 4;
  const int h = bh & 15;
#pragma unroll
  for (int v = 0; v < 8; ++v) {
    const float inv = 1.0f / lrow[v];
    const int trow  = q0 + v + 8 * lg;
    const size_t ybase = ((size_t)(b * T_ + trow)) * C_ + h * HD_;
#pragma unroll
    for (int nc = 0; nc < 8; ++nc)
      yb[ybase + nc * 16 + lm] = (bf16_t)(O[nc][v] * inv);
  }
}

// ---------------------------------------------------------------------------
extern "C" void kernel_launch(void* const* d_in, const int* in_sizes, int n_in,
                              void* d_out, int out_size, void* d_ws,
                              size_t ws_size, hipStream_t stream) {
  const float* x      = (const float*)d_in[0];  // [B,T,C]
  const float* cosb   = (const float*)d_in[1];  // [T,64]
  const float* sinb   = (const float*)d_in[2];  // [T,64]
  const float* W_attn = (const float*)d_in[3];  // [C,3C]
  const float* W_proj = (const float*)d_in[4];  // [C,C]
  float* out = (float*)d_out;                   // [B,T,C]

  char* ws = (char*)d_ws;
  float*  qkv = (float*)ws;                                  // 4096*6144 f32
  bf16_t* qd  = (bf16_t*)(ws + (size_t)4096 * 6144 * 4);     // 8388608 bf16
  bf16_t* kd  = qd + (size_t)8388608;
  bf16_t* vT  = kd + (size_t)8388608;
  bf16_t* yb  = vT + (size_t)8388608;

  // 1) QKV projection: [4096,2048] @ [2048,6144] -> qkv fp32
  gemm_wmma_bf16<float>
      <<<dim3(C3_ / 128, (B_ * T_) / 128), 256, 0, stream>>>(
          x, W_attn, qkv, B_ * T_, C3_, C_);

  // 2) RoPE + head split (+ q pre-scale, v transpose)
  rope_split_kernel<<<(B_ * T_ * H_) / 256, 256, 0, stream>>>(
      qkv, cosb, sinb, qd, kd, vT);

  // 3) causal flash attention: one wave per (b,h,q-tile)
  attn_kernel<<<B_ * H_ * (T_ / 16), 32, 0, stream>>>(qd, kd, vT, yb);

  // 4) output projection: [4096,2048](bf16) @ [2048,2048] -> out fp32
  gemm_wmma_bf16<bf16_t>
      <<<dim3(C_ / 128, (B_ * T_) / 128), 256, 0, stream>>>(
          yb, W_proj, out, B_ * T_, C_, C_);
}
